// TargetNet_21552145891466
// MI455X (gfx1250) — compile-verified, matrix operand order checked by reading, not run on hardware
//
#include <hip/hip_runtime.h>

#define SD      2000000
#define KTILE   32
#define NTILES  (SD / KTILE)      // 62500, exact
#define NBLK    512
#define NTHR    256

typedef __attribute__((ext_vector_type(16))) __bf16 v16bf;
typedef __attribute__((ext_vector_type(8)))  float  v8f;
typedef __attribute__((ext_vector_type(4)))  float  f4;

__device__ __forceinline__ f4 ldnt4(const float* p) {
    return __builtin_nontemporal_load((const f4*)p);
}

// d_ws layout (floats): [0..31] = h accumulators, [32..63] = rowsum accumulators
__global__ void init_acc_kernel(float* acc) {
    int i = threadIdx.x;
    if (i < 64) acc[i] = 0.0f;
}

__global__ void __launch_bounds__(NTHR)
matvec_wmma_kernel(const float* __restrict__ W1, const float* __restrict__ x,
                   float* __restrict__ acc) {
    __shared__ float sh[64];
    const int tid = threadIdx.x;
    if (tid < 64) sh[tid] = 0.0f;
    __syncthreads();

    const int lane = tid & 31;
    const int half = lane >> 4;   // selects K sub-group (ISA 16-bit A layout)
    const int m    = lane & 15;   // row within tile / column id in B & C

    const int waveId   = blockIdx.x * (NTHR >> 5) + (tid >> 5);
    const int numWaves = NBLK * (NTHR >> 5);
    const int per = (NTILES + numWaves - 1) / numWaves;
    int t0 = waveId * per;
    int t1 = t0 + per;
    if (t0 > NTILES) t0 = NTILES;
    if (t1 > NTILES) t1 = NTILES;

    const bool isX   = (m == 0);   // B column 0 carries x
    const bool isOne = (m == 1);   // B column 1 carries ones (row-sum)

    v8f c0 = {};   // rows 0..15  x 16 cols
    v8f c1 = {};   // rows 16..31 x 16 cols

    const long rowA = (long)m * SD;          // tile0 row base
    const long rowB = (long)(m + 16) * SD;   // tile1 row base

    for (int t = t0; t < t1; ++t) {
        const long koff = (long)t * KTILE + half * 8;

        // A fragments: 8 contiguous K at koff (elems 0..7) + 8 at koff+16 (elems 8..15)
        const float* p0 = W1 + rowA + koff;
        const float* p1 = W1 + rowB + koff;
        f4 a00 = ldnt4(p0),      a01 = ldnt4(p0 + 4);
        f4 a02 = ldnt4(p0 + 16), a03 = ldnt4(p0 + 20);
        f4 a10 = ldnt4(p1),      a11 = ldnt4(p1 + 4);
        f4 a12 = ldnt4(p1 + 16), a13 = ldnt4(p1 + 20);

        // B fragment sources (only lanes with N==0 touch x)
        f4 xb0 = {0.f,0.f,0.f,0.f}, xb1 = xb0, xb2 = xb0, xb3 = xb0;
        if (isX) {
            const float* xp = x + koff;
            xb0 = ldnt4(xp);      xb1 = ldnt4(xp + 4);
            xb2 = ldnt4(xp + 16); xb3 = ldnt4(xp + 20);
        }

        v16bf a0, a1, b;
        #pragma unroll
        for (int i = 0; i < 4; ++i) {
            a0[i]      = (__bf16)a00[i];
            a0[4 + i]  = (__bf16)a01[i];
            a0[8 + i]  = (__bf16)a02[i];
            a0[12 + i] = (__bf16)a03[i];
            a1[i]      = (__bf16)a10[i];
            a1[4 + i]  = (__bf16)a11[i];
            a1[8 + i]  = (__bf16)a12[i];
            a1[12 + i] = (__bf16)a13[i];
            const float one0 = isOne ? 1.0f : 0.0f;
            b[i]      = (__bf16)(isX ? xb0[i] : one0);
            b[4 + i]  = (__bf16)(isX ? xb1[i] : one0);
            b[8 + i]  = (__bf16)(isX ? xb2[i] : one0);
            b[12 + i] = (__bf16)(isX ? xb3[i] : one0);
        }

        // D = A x B + C, f32 accumulate (8 args: neg_a, A, neg_b, B, c_mod, C, reuse)
        c0 = __builtin_amdgcn_wmma_f32_16x16x32_bf16(false, a0, false, b,
                                                     (short)0, c0, false, false);
        c1 = __builtin_amdgcn_wmma_f32_16x16x32_bf16(false, a1, false, b,
                                                     (short)0, c1, false, false);
    }

    // C layout: VGPR r holds (M=r, N=lane) for lanes 0-15 and (M=8+r, N=lane-16)
    // for lanes 16-31. Column 0 (h) lives in lanes 0/16; column 1 (rowsum) in 1/17.
    const int mbase = half * 8;
    if (m == 0) {
        #pragma unroll
        for (int r = 0; r < 8; ++r) {
            atomicAdd(&sh[mbase + r],      c0[r]);   // h rows 0..15
            atomicAdd(&sh[16 + mbase + r], c1[r]);   // h rows 16..31
        }
    } else if (m == 1) {
        #pragma unroll
        for (int r = 0; r < 8; ++r) {
            atomicAdd(&sh[32 + mbase + r],      c0[r]);  // rowsum rows 0..15
            atomicAdd(&sh[32 + 16 + mbase + r], c1[r]);  // rowsum rows 16..31
        }
    }
    __syncthreads();
    if (tid < 64) atomicAdd(&acc[tid], sh[tid]);
}

__global__ void finalize_kernel(const float* __restrict__ acc,
                                const float* __restrict__ b1,
                                const float* __restrict__ W2,
                                const float* __restrict__ b2,
                                float* __restrict__ out) {
    const int i = threadIdx.x;          // one wave32, i = 0..31
    const float h  = acc[i] + b1[i];
    const float rs = acc[32 + i];
    // s_i = #{W1[i,:] == -1} + (b1_i == -1) ; rowsum = K - 2*s  =>  s = (K - rowsum)/2
    const float s = ((float)SD - rs) * 0.5f + (b1[i] == -1.0f ? 1.0f : 0.0f);
    const float theta = (float)(SD + 1) * 0.7f - s;
    const float hb = (h > theta) ? 1.0f : 0.0f;
    float v = W2[i] * hb;
    #pragma unroll
    for (int off = 16; off > 0; off >>= 1)
        v += __shfl_down(v, off, 32);
    if (i == 0) out[0] = v + b2[0];
}

extern "C" void kernel_launch(void* const* d_in, const int* in_sizes, int n_in,
                              void* d_out, int out_size, void* d_ws, size_t ws_size,
                              hipStream_t stream) {
    const float* x  = (const float*)d_in[0];
    const float* W1 = (const float*)d_in[1];
    const float* b1 = (const float*)d_in[2];
    const float* W2 = (const float*)d_in[3];
    const float* b2 = (const float*)d_in[4];
    float* out = (float*)d_out;
    float* acc = (float*)d_ws;   // 64 floats of scratch

    init_acc_kernel<<<1, 64, 0, stream>>>(acc);
    matvec_wmma_kernel<<<NBLK, NTHR, 0, stream>>>(W1, x, acc);
    finalize_kernel<<<1, 32, 0, stream>>>(acc, b1, W2, b2, out);
}
